// LSTM_79413945303609
// MI455X (gfx1250) — compile-verified, matrix operand order checked by reading, not run on hardware
//
#include <hip/hip_runtime.h>
#include <hip/hip_bf16.h>

// Problem constants: B=128, T=256, D=512, H=1024, O=1024
#define BB   128
#define TT   256
#define DD   512
#define HH   1024
#define OO   1024
#define KK   (DD + HH)     // 1536 fused K (x | h)
#define NN   (4 * HH)      // 4096 gate columns

typedef __attribute__((ext_vector_type(8)))  float   v8f;
typedef __attribute__((ext_vector_type(4)))  __bf16  v4bf;
typedef __attribute__((ext_vector_type(8)))  __bf16  v8bf;
typedef __attribute__((ext_vector_type(16))) __bf16  v16bf;

// ---------------- workspace layout (bytes) ----------------
// W_cat bf16 [NN][KK]      : 4096*1536*2 = 12,582,912
// A     bf16 [2][BB][KK]   : 2*128*1536*2 =   786,432   (double-buffered x|h)
// h     f32  [BB][HH]      : 524,288
// c     f32  [BB][HH]      : 524,288
// sync  int  [2]
#define OFF_W 0
#define OFF_A (OFF_W + (size_t)NN * KK * 2)
#define OFF_H (OFF_A + (size_t)2 * BB * KK * 2)
#define OFF_C (OFF_H + (size_t)BB * HH * 4)
#define OFF_S (OFF_C + (size_t)BB * HH * 4)

// ---------------- prep: pack weights to bf16 K-major, init state ----------------
__global__ void lstm_prep(const float* __restrict__ x,
                          const float* __restrict__ Wx,
                          const float* __restrict__ Wh,
                          __bf16* __restrict__ W,
                          __bf16* __restrict__ A,
                          float* __restrict__ h,
                          float* __restrict__ c,
                          int* __restrict__ sync) {
    const long tid    = (long)blockIdx.x * blockDim.x + threadIdx.x;
    const long stride = (long)gridDim.x * blockDim.x;

    // W_cat[n][k]: n = g*H + o ; k<D -> Wx[g][o][k], else Wh[g][o][k-D]
    for (long i = tid; i < (long)NN * KK; i += stride) {
        const int n = (int)(i / KK), k = (int)(i % KK);
        const int g = n >> 10, o = n & (HH - 1);
        float v;
        if (k < DD) v = Wx[((long)(g * HH + o)) * DD + k];
        else        v = Wh[((long)(g * HH + o)) * HH + (k - DD)];
        W[i] = (__bf16)v;
    }
    // A bank 0: x part = x[:,0,:], h part = 0
    for (long i = tid; i < (long)BB * KK; i += stride) {
        const int b = (int)(i / KK), k = (int)(i % KK);
        A[i] = (k < DD) ? (__bf16)x[((long)b * TT) * DD + k] : (__bf16)0.0f;
    }
    // h, c = 0
    for (long i = tid; i < (long)BB * HH; i += stride) { h[i] = 0.0f; c[i] = 0.0f; }
    if (tid < 2) sync[tid] = 0;
}

// ---------------- grid-wide barrier (persistent kernel) ----------------
__device__ inline void grid_sync(int* cnt, int* gen, int nblocks) {
    __syncthreads();
    if (threadIdx.x == 0) {
        __threadfence();
        const int my = __hip_atomic_load(gen, __ATOMIC_RELAXED, __HIP_MEMORY_SCOPE_AGENT);
        const int prev = __hip_atomic_fetch_add(cnt, 1, __ATOMIC_ACQ_REL, __HIP_MEMORY_SCOPE_AGENT);
        if (prev == nblocks - 1) {
            __hip_atomic_store(cnt, 0, __ATOMIC_RELAXED, __HIP_MEMORY_SCOPE_AGENT);
            __hip_atomic_store(gen, my + 1, __ATOMIC_RELEASE, __HIP_MEMORY_SCOPE_AGENT);
        } else {
            while (__hip_atomic_load(gen, __ATOMIC_ACQUIRE, __HIP_MEMORY_SCOPE_AGENT) == my) {
                __builtin_amdgcn_s_sleep(2);
            }
        }
        __threadfence();
    }
    __syncthreads();
}

__device__ inline float sigmoidf(float v) { return 1.0f / (1.0f + __expf(-v)); }
__device__ inline float tanh_fast(float v) { return 2.0f * sigmoidf(2.0f * v) - 1.0f; }

__device__ inline v16bf ldfrag(const __bf16* p) {
    const v8bf lo = *(const v8bf*)(p);
    const v8bf hi = *(const v8bf*)(p + 16);
    return __builtin_shufflevector(lo, hi, 0,1,2,3,4,5,6,7,8,9,10,11,12,13,14,15);
}

// ---------------- persistent recurrent kernel ----------------
// Grid: 128 blocks x 128 threads = 512 waves; each wave owns one 16(M)x16(o)
// patch across ALL FOUR gates -> in-register cell update, ONE barrier/step.
__global__ __launch_bounds__(128) void lstm_persistent(
        const float* __restrict__ x,      // [B,T,D] f32
        const float* __restrict__ bias,   // [4,H] f32
        const __bf16* __restrict__ W,     // [NN][KK] bf16, K-major
        __bf16* __restrict__ A,           // [2][BB][KK] bf16 (x_t | h), double-buffered
        float* __restrict__ h,
        float* __restrict__ c,
        int* __restrict__ sync) {
    const int nblocks = gridDim.x;
    const int wave    = (blockIdx.x << 2) + (threadIdx.x >> 5);   // 0..511
    const int lane    = threadIdx.x & 31;
    const int lane15  = lane & 15;
    const int half8   = (lane >> 4) << 3;        // 0|8: K-half (A/B frags), M-offset (C)
    const int mt      = wave & 7;                // 8 M-tiles of 16 rows
    const int ot      = wave >> 3;               // 64 o-tiles of 16 cols
    const int ocol    = (ot << 4) + lane15;      // this lane's o column
    const int mrow    = (mt << 4) + half8;       // this lane's first M row in C
    int* cnt = sync;
    int* gen = sync + 1;

    // per-lane gate biases for column ocol
    const float bi = bias[ocol];
    const float bf_ = bias[HH + ocol];
    const float bg = bias[2 * HH + ocol];
    const float bo = bias[3 * HH + ocol];

    // B-operand row pointers, one per gate (rows n = g*H + ot*16 + lane15)
    const __bf16* Bp0 = W + ((size_t)(0 * HH + (ot << 4) + lane15)) * KK + half8;
    const __bf16* Bp1 = Bp0 + (size_t)HH * KK;
    const __bf16* Bp2 = Bp1 + (size_t)HH * KK;
    const __bf16* Bp3 = Bp2 + (size_t)HH * KK;

    for (int t = 0; t < TT; ++t) {
        const __bf16* Acur  = A + (size_t)(t & 1) * BB * KK;
        __bf16*       Anext = A + (size_t)((t + 1) & 1) * BB * KK;
        const __bf16* Ap    = Acur + (size_t)((mt << 4) + lane15) * KK + half8;

        v8f acc0 = {}, acc1 = {}, acc2 = {}, acc3 = {};
        for (int k = 0; k < KK; k += 32) {
            __builtin_prefetch(Bp0 + k + 256, 0, 1);
            __builtin_prefetch(Bp1 + k + 256, 0, 1);
            __builtin_prefetch(Bp2 + k + 256, 0, 1);
            __builtin_prefetch(Bp3 + k + 256, 0, 1);
            const v16bf av  = ldfrag(Ap + k);     // reused by all 4 gates
            const v16bf bv0 = ldfrag(Bp0 + k);
            const v16bf bv1 = ldfrag(Bp1 + k);
            const v16bf bv2 = ldfrag(Bp2 + k);
            const v16bf bv3 = ldfrag(Bp3 + k);
            // four INDEPENDENT accumulators -> no WMMA->WMMA RAW stalls
            acc0 = __builtin_amdgcn_wmma_f32_16x16x32_bf16(false, av, false, bv0, (short)0, acc0, false, false);
            acc1 = __builtin_amdgcn_wmma_f32_16x16x32_bf16(false, av, false, bv1, (short)0, acc1, false, false);
            acc2 = __builtin_amdgcn_wmma_f32_16x16x32_bf16(false, av, false, bv2, (short)0, acc2, false, false);
            acc3 = __builtin_amdgcn_wmma_f32_16x16x32_bf16(false, av, false, bv3, (short)0, acc3, false, false);
        }

        // ---- in-register cell update: lane holds i/f/g/o for 8 (row, ocol) elems ----
        #pragma unroll
        for (int r = 0; r < 8; ++r) {
            const int row = mrow + r;
            const int idx = row * HH + ocol;
            const float gi = sigmoidf(acc0[r] + bi);
            const float gf = sigmoidf(acc1[r] + bf_);
            const float gg = tanh_fast(acc2[r] + bg);
            const float go = sigmoidf(acc3[r] + bo);
            const float cv = gf * c[idx] + gi * gg;   // c owned by this wave only
            c[idx] = cv;
            const float hv = go * tanh_fast(cv);
            Anext[(size_t)row * KK + DD + ocol] = (__bf16)hv;
            if (t == TT - 1) h[idx] = hv;             // fp32 h only needed at the end
        }

        // ---- refill x part of NEXT bank (64 waves per m-tile split the 16x512 slab) ----
        if (t + 1 < TT) {
            const int flat = (ot << 7) + (lane << 2);        // 0..8191, 4 elems/lane
            const int r = flat >> 9, d = flat & (DD - 1);
            const int b = (mt << 4) + r;
            const float4 xv = *(const float4*)(x + ((size_t)b * TT + (t + 1)) * DD + d);
            const v4bf xb = { (__bf16)xv.x, (__bf16)xv.y, (__bf16)xv.z, (__bf16)xv.w };
            *(v4bf*)(Anext + (size_t)b * KK + d) = xb;
        }

        grid_sync(cnt, gen, nblocks);   // single barrier per step
    }
}

// ---------------- head: logits = h @ Whp^T + bhp ; softmax over O ----------------
__global__ __launch_bounds__(256) void lstm_head(const float* __restrict__ h,
                                                 const float* __restrict__ Whp,
                                                 const float* __restrict__ bhp,
                                                 float* __restrict__ out) {
    const int b = blockIdx.x;
    __shared__ float sl[OO];
    __shared__ float red[256];
    const float* hrow = h + (size_t)b * HH;

    for (int o = threadIdx.x; o < OO; o += blockDim.x) {
        const float* w = Whp + (size_t)o * HH;
        float acc = bhp[o];
        #pragma unroll 4
        for (int k = 0; k < HH; ++k) acc = fmaf(hrow[k], w[k], acc);
        sl[o] = acc;
    }
    __syncthreads();

    float m = -3.4e38f;
    for (int o = threadIdx.x; o < OO; o += blockDim.x) m = fmaxf(m, sl[o]);
    red[threadIdx.x] = m;
    __syncthreads();
    for (int s = 128; s > 0; s >>= 1) {
        if (threadIdx.x < s) red[threadIdx.x] = fmaxf(red[threadIdx.x], red[threadIdx.x + s]);
        __syncthreads();
    }
    m = red[0];
    __syncthreads();

    float p = 0.0f;
    for (int o = threadIdx.x; o < OO; o += blockDim.x) {
        const float e = __expf(sl[o] - m);
        sl[o] = e;
        p += e;
    }
    red[threadIdx.x] = p;
    __syncthreads();
    for (int s = 128; s > 0; s >>= 1) {
        if (threadIdx.x < s) red[threadIdx.x] += red[threadIdx.x + s];
        __syncthreads();
    }
    const float inv = 1.0f / red[0];
    for (int o = threadIdx.x; o < OO; o += blockDim.x) out[(size_t)b * OO + o] = sl[o] * inv;
}

extern "C" void kernel_launch(void* const* d_in, const int* in_sizes, int n_in,
                              void* d_out, int out_size, void* d_ws, size_t ws_size,
                              hipStream_t stream) {
    const float* x   = (const float*)d_in[0];
    const float* Wx  = (const float*)d_in[1];
    const float* Wh  = (const float*)d_in[2];
    const float* b   = (const float*)d_in[3];
    const float* Whp = (const float*)d_in[4];
    const float* bhp = (const float*)d_in[5];
    float* out = (float*)d_out;

    char* ws = (char*)d_ws;
    __bf16* W  = (__bf16*)(ws + OFF_W);
    __bf16* A  = (__bf16*)(ws + OFF_A);
    float* h   = (float*)(ws + OFF_H);
    float* c   = (float*)(ws + OFF_C);
    int*   sync = (int*)(ws + OFF_S);

    lstm_prep<<<1024, 256, 0, stream>>>(x, Wx, Wh, W, A, h, c, sync);
    lstm_persistent<<<128, 128, 0, stream>>>(x, b, W, A, h, c, sync);
    lstm_head<<<BB, 256, 0, stream>>>(h, Whp, bhp, out);
}
